// GATClassifierWithMLP_55259049230617
// MI455X (gfx1250) — matrix operand order repaired and checked by reading, not compiled, not run on hardware
//
#include <hip/hip_runtime.h>
#include <hip/hip_bf16.h>

typedef __attribute__((ext_vector_type(16))) _Float16 v16h;
typedef __attribute__((ext_vector_type(8)))  _Float16 v8h;
typedef __attribute__((ext_vector_type(8)))  float    v8f;

#define NN 50000
#define NE 800000
#define INDIM 512
#define HD 256            // HEADS*HID
#define NCLS 10

// ---------- helpers ----------
__device__ __forceinline__ unsigned fenc(float f) {
    unsigned u = __float_as_uint(f);
    return (u & 0x80000000u) ? ~u : (u | 0x80000000u);
}
__device__ __forceinline__ float fdec(unsigned u) {
    return __uint_as_float((u & 0x80000000u) ? (u ^ 0x80000000u) : ~u);
}

// async copy 16B: global -> LDS, tracked by ASYNCcnt (CDNA5 GLOBAL_LOAD_ASYNC_TO_LDS_B128)
__device__ __forceinline__ void async_g2l_b128(unsigned lds_addr, const void* gaddr) {
    asm volatile("global_load_async_to_lds_b128 %0, %1, off"
                 :: "v"(lds_addr), "v"(gaddr) : "memory");
}
__device__ __forceinline__ void wait_asynccnt0() {
    asm volatile("s_wait_asynccnt 0x0" ::: "memory");
}

// ---------- f32 -> f16 convert (activations) ----------
__global__ void cvt_f32_f16(const float* __restrict__ in, _Float16* __restrict__ out, int n) {
    int i = blockIdx.x * 256 + threadIdx.x;
    if (i < n) out[i] = (_Float16)in[i];
}

// ---------- f32 [K,N] -> f16 transposed [N,K] (weights) ----------
__global__ void cvt_w_t(const float* __restrict__ in, _Float16* __restrict__ out,
                        int K, int N) {
    int i = blockIdx.x * 256 + threadIdx.x;
    if (i >= K * N) return;
    int k = i / N, n = i % N;
    out[(size_t)n * K + k] = (_Float16)in[i];
}

// ---------- WMMA GEMM: C[M,N] = A[M,K](f16,row) @ Bt[N,K](f16,row = B^T) ----------
// 128x128 block tile, BK=32, double-buffered LDS with async global->LDS staging,
// 8 waves each owning 32(M)x64(N) = 8 WMMAs per K-step.
__global__ __launch_bounds__(256)
void gemm_wmma_f16(const _Float16* __restrict__ A, const _Float16* __restrict__ Bt,
                   float* __restrict__ C, _Float16* __restrict__ C16,
                   const float* __restrict__ bias, int M, int N, int K, int doRelu)
{
    const int LD = 40;                              // padded f16 stride (80B, 16B-aligned)
    __shared__ _Float16 lA[2][128 * 40];
    __shared__ _Float16 lB[2][128 * 40];

    const int tid    = threadIdx.x;
    const int wid    = tid >> 5;
    const int lane   = tid & 31;
    const int lane16 = lane & 15;
    const int half   = lane >> 4;
    const int waveM  = wid & 3;                     // 4 row-waves -> 32 rows each
    const int waveN  = wid >> 2;                    // 2 col-waves -> 64 cols each
    const int blockRow = blockIdx.x * 128;
    const int blockCol = blockIdx.y * 128;

    // async-stage one BK=32 slab of A (128 rows) and Bt (128 cols) into buffer `buf`
    auto stage = [&](int buf, int k0) {
        #pragma unroll
        for (int it = 0; it < 2; ++it) {
            int s   = tid + it * 256;               // 512 slots of 8 halfs
            int row = s >> 2;
            int c8  = (s & 3) * 8;
            int gr  = blockRow + row; if (gr >= M) gr = M - 1;   // clamp; store guarded
            async_g2l_b128((unsigned)(size_t)&lA[buf][row * LD + c8],
                           A + (size_t)gr * K + k0 + c8);
            int gn  = blockCol + row;               // N is a multiple of 128: in range
            async_g2l_b128((unsigned)(size_t)&lB[buf][row * LD + c8],
                           Bt + (size_t)gn * K + k0 + c8);
        }
    };

    v8f acc[2][4] = {};
    const int nk = K >> 5;

    stage(0, 0);
    wait_asynccnt0();
    __syncthreads();

    for (int kk = 0; kk < nk; ++kk) {
        int cur = kk & 1;
        if (kk + 1 < nk) stage(cur ^ 1, (kk + 1) * 32);   // async prefetch next slab

        // fragments (ISA 7.12.2): A lane=M row, elems 0-7:K=half*8+e, 8-15:K=16+half*8+e
        //                         B lane=N col, elems 0-7:K=half*8+e, 8-15:K=16+half*8+e
        union { v16h v; v8h h[2]; } af[2], bf[4];
        #pragma unroll
        for (int i = 0; i < 2; ++i) {
            const _Float16* base = &lA[cur][(waveM * 32 + i * 16 + lane16) * LD];
            af[i].h[0] = *(const v8h*)(base + half * 8);
            af[i].h[1] = *(const v8h*)(base + 16 + half * 8);
        }
        #pragma unroll
        for (int j = 0; j < 4; ++j) {
            const _Float16* base = &lB[cur][(waveN * 64 + j * 16 + lane16) * LD];
            bf[j].h[0] = *(const v8h*)(base + half * 8);
            bf[j].h[1] = *(const v8h*)(base + 16 + half * 8);
        }
        #pragma unroll
        for (int i = 0; i < 2; ++i)
            #pragma unroll
            for (int j = 0; j < 4; ++j)
                acc[i][j] = __builtin_amdgcn_wmma_f32_16x16x32_f16(
                    false, af[i].v, false, bf[j].v, (short)0, acc[i][j], false, false);

        wait_asynccnt0();                           // our async writes done before barrier
        __syncthreads();
    }

    // epilogue: C/D layout -> row = tile + r + 8*half, col = tile + lane16
    #pragma unroll
    for (int i = 0; i < 2; ++i) {
        #pragma unroll
        for (int j = 0; j < 4; ++j) {
            int col = blockCol + waveN * 64 + j * 16 + lane16;
            #pragma unroll
            for (int r = 0; r < 8; ++r) {
                int row = blockRow + waveM * 32 + i * 16 + half * 8 + r;
                if (row < M) {
                    float v = acc[i][j][r];
                    if (bias)   v += bias[col];
                    if (doRelu) v  = fmaxf(v, 0.0f);
                    C[(size_t)row * N + col] = v;
                    if (C16) C16[(size_t)row * N + col] = (_Float16)v;
                }
            }
        }
    }
}

// ---------- el/er = einsum('nhd,hd->nh') ----------
__global__ void attn_dots(const float* __restrict__ h, const float* __restrict__ al,
                          const float* __restrict__ ar, float* __restrict__ el,
                          float* __restrict__ er, int n)
{
    int i = blockIdx.x * 256 + threadIdx.x;       // over n*4
    if (i >= n * 4) return;
    int node = i >> 2, hd = i & 3;
    const float4* hp = (const float4*)(h + (size_t)node * HD + hd * 64);
    const float4* lp = (const float4*)(al + hd * 64);
    const float4* rp = (const float4*)(ar + hd * 64);
    float sl = 0.f, sr = 0.f;
    #pragma unroll 4
    for (int q = 0; q < 16; ++q) {
        float4 hv = hp[q], lv = lp[q], rv = rp[q];
        sl += hv.x * lv.x + hv.y * lv.y + hv.z * lv.z + hv.w * lv.w;
        sr += hv.x * rv.x + hv.y * rv.y + hv.z * rv.z + hv.w * rv.w;
    }
    el[i] = sl; er[i] = sr;
}

// ---------- per-edge leaky-relu + segment max (atomic on monotone uint encoding) ----------
__global__ void edge_max(const int* __restrict__ src, const int* __restrict__ dst,
                         const float* __restrict__ el, const float* __restrict__ er,
                         float* __restrict__ eedge, unsigned* __restrict__ menc, int ne)
{
    int e = blockIdx.x * 256 + threadIdx.x;
    if (e >= ne) return;
    int s = src[e], d = dst[e];
    float4 ls = *(const float4*)(el + (size_t)s * 4);
    float4 rd = *(const float4*)(er + (size_t)d * 4);
    float v[4] = { ls.x + rd.x, ls.y + rd.y, ls.z + rd.z, ls.w + rd.w };
    #pragma unroll
    for (int hd = 0; hd < 4; ++hd) {
        float x = v[hd];
        x = x > 0.f ? x : 0.2f * x;               // leaky relu, slope 0.2
        v[hd] = x;
        atomicMax(&menc[d * 4 + hd], fenc(x));
    }
    float4 o; o.x = v[0]; o.y = v[1]; o.z = v[2]; o.w = v[3];
    *(float4*)(eedge + (size_t)e * 4) = o;
}

// ---------- one wave per edge: ex = exp(e-m); accumulate denom and ex * h_src ----------
__global__ void edge_sum(const int* __restrict__ src, const int* __restrict__ dst,
                         const float* __restrict__ eedge, const unsigned* __restrict__ menc,
                         const float* __restrict__ h, float* __restrict__ out,
                         float* __restrict__ denom, int ne)
{
    int gt = blockIdx.x * 256 + threadIdx.x;
    int e  = gt >> 5;                             // one wave32 per edge
    if (e >= ne) return;                          // uniform per wave
    int lane = gt & 31;
    int s = src[e], d = dst[e];
    int hd = lane >> 3;
    float m  = fdec(menc[d * 4 + hd]);
    float ex = __expf(eedge[(size_t)e * 4 + hd] - m);
    if ((lane & 7) == 0) atomicAdd(&denom[d * 4 + hd], ex);
    const float* hs = h   + (size_t)s * HD + lane * 8;
    float*       op = out + (size_t)d * HD + lane * 8;
    float4 h0 = *(const float4*)(hs);
    float4 h1 = *(const float4*)(hs + 4);
    atomicAdd(op + 0, ex * h0.x); atomicAdd(op + 1, ex * h0.y);
    atomicAdd(op + 2, ex * h0.z); atomicAdd(op + 3, ex * h0.w);
    atomicAdd(op + 4, ex * h1.x); atomicAdd(op + 5, ex * h1.y);
    atomicAdd(op + 6, ex * h1.z); atomicAdd(op + 7, ex * h1.w);
}

// ---------- out = relu(sum/denom + bias); also emit f16 copy for next GEMM ----------
__global__ void finalize_layer(float* __restrict__ buf, const float* __restrict__ denom,
                               const float* __restrict__ bias, _Float16* __restrict__ out16,
                               int n)
{
    int i = blockIdx.x * 256 + threadIdx.x;       // over n*256
    if (i >= n * HD) return;
    int node = i >> 8, t = i & 255;
    float dn = denom[node * 4 + (t >> 6)];
    float v  = buf[i] / fmaxf(dn, 1e-9f) + bias[t];
    v = fmaxf(v, 0.0f);
    buf[i]   = v;
    out16[i] = (_Float16)v;
}

// ---------- classifier: [N,256] @ [256,10] + b ----------
__global__ void clf_kernel(const float* __restrict__ x, const float* __restrict__ w,
                           const float* __restrict__ b, float* __restrict__ out, int n)
{
    int i = blockIdx.x * 256 + threadIdx.x;
    if (i >= n * NCLS) return;
    int node = i / NCLS, c = i % NCLS;
    const float* xr = x + (size_t)node * HD;
    float s = b[c];
    #pragma unroll 8
    for (int k = 0; k < HD; ++k) s += xr[k] * w[k * NCLS + c];
    out[i] = s;
}

// ==================== launch ====================
extern "C" void kernel_launch(void* const* d_in, const int* in_sizes, int n_in,
                              void* d_out, int out_size, void* d_ws, size_t ws_size,
                              hipStream_t stream)
{
    const float* features = (const float*)d_in[0];
    const int*   src      = (const int*)  d_in[1];
    const int*   dst      = (const int*)  d_in[2];
    const float* W1       = (const float*)d_in[3];
    const float* al1      = (const float*)d_in[4];
    const float* ar1      = (const float*)d_in[5];
    const float* b1       = (const float*)d_in[6];
    const float* W2       = (const float*)d_in[7];
    const float* al2      = (const float*)d_in[8];
    const float* ar2      = (const float*)d_in[9];
    const float* b2       = (const float*)d_in[10];
    const float* fc1_w    = (const float*)d_in[11];
    const float* fc1_b    = (const float*)d_in[12];
    const float* fc2_w    = (const float*)d_in[13];
    const float* fc2_b    = (const float*)d_in[14];
    const float* clf_w    = (const float*)d_in[15];
    const float* clf_b    = (const float*)d_in[16];
    float* out = (float*)d_out;

    // workspace carve-out (256B aligned)
    char*  ws  = (char*)d_ws;
    size_t off = 0;
    auto carve = [&](size_t bytes) -> void* {
        off = (off + 255) & ~(size_t)255;
        void* p = ws + off;
        off += bytes;
        return p;
    };
    float*     A32   = (float*)    carve((size_t)NN * 512 * 4);   // GEMM outputs (pre-agg h, fc1)
    float*     B32   = (float*)    carve((size_t)NN * HD  * 4);   // aggregated / fc2 out
    _Float16*  H16a  = (_Float16*) carve((size_t)NN * 512 * 2);   // f16 activations (GEMM input)
    _Float16*  H16b  = (_Float16*) carve((size_t)NN * 512 * 2);   // f16 fc1 output
    _Float16*  W16T  = (_Float16*) carve((size_t)512 * 512 * 2);  // f16 transposed weights
    float*     eedge = (float*)    carve((size_t)NE * 4 * 4);
    float*     el    = (float*)    carve((size_t)NN * 4 * 4);
    float*     er    = (float*)    carve((size_t)NN * 4 * 4);
    unsigned*  menc  = (unsigned*) carve((size_t)NN * 4 * 4);
    float*     denom = (float*)    carve((size_t)NN * 4 * 4);

    auto blocks = [](long n) { return (int)((n + 255) / 256); };
    const dim3 blk(256);

    // one GAT layer: H16a[N,K] -> B32[N,256] (f32) and H16a[N,256] (f16)
    auto gat_layer = [&](const float* W, int K, const float* al, const float* ar,
                         const float* bias) {
        cvt_w_t<<<blocks((long)K * HD), blk, 0, stream>>>(W, W16T, K, HD);
        dim3 grid((NN + 127) / 128, HD / 128);
        gemm_wmma_f16<<<grid, blk, 0, stream>>>(H16a, W16T, A32, nullptr, nullptr,
                                                NN, HD, K, 0);
        attn_dots<<<blocks((long)NN * 4), blk, 0, stream>>>(A32, al, ar, el, er, NN);
        hipMemsetAsync(menc,  0, (size_t)NN * 4 * 4, stream);
        hipMemsetAsync(denom, 0, (size_t)NN * 4 * 4, stream);
        hipMemsetAsync(B32,   0, (size_t)NN * HD * 4, stream);
        edge_max<<<blocks(NE), blk, 0, stream>>>(src, dst, el, er, eedge, menc, NE);
        edge_sum<<<blocks((long)NE * 32), blk, 0, stream>>>(src, dst, eedge, menc,
                                                            A32, B32, denom, NE);
        finalize_layer<<<blocks((long)NN * HD), blk, 0, stream>>>(B32, denom, bias,
                                                                  H16a, NN);
    };

    // features -> f16
    cvt_f32_f16<<<blocks((long)NN * INDIM), blk, 0, stream>>>(features, H16a, NN * INDIM);

    gat_layer(W1, INDIM, al1, ar1, b1);   // layer 1: K=512
    gat_layer(W2, HD,    al2, ar2, b2);   // layer 2: K=256

    // fc1: [N,256] @ [256,512], relu, f16 copy into H16b
    cvt_w_t<<<blocks((long)HD * 512), blk, 0, stream>>>(fc1_w, W16T, HD, 512);
    {
        dim3 grid((NN + 127) / 128, 512 / 128);
        gemm_wmma_f16<<<grid, blk, 0, stream>>>(H16a, W16T, A32, H16b, fc1_b,
                                                NN, 512, HD, 1);
    }
    // fc2: [N,512] @ [512,256], relu
    cvt_w_t<<<blocks((long)512 * HD), blk, 0, stream>>>(fc2_w, W16T, 512, HD);
    {
        dim3 grid((NN + 127) / 128, HD / 128);
        gemm_wmma_f16<<<grid, blk, 0, stream>>>(H16b, W16T, B32, nullptr, fc2_b,
                                                NN, HD, 512, 1);
    }
    // classifier
    clf_kernel<<<blocks((long)NN * NCLS), blk, 0, stream>>>(B32, clf_w, clf_b, out, NN);
}
